// DSCA_326417515012
// MI455X (gfx1250) — matrix operand order repaired and checked by reading, not compiled, not run on hardware
//
#include <hip/hip_runtime.h>

// ---------- types ----------
typedef unsigned short u16;
typedef unsigned int   u32;
typedef __attribute__((ext_vector_type(8)))  u16    u16x8;
typedef __attribute__((ext_vector_type(16))) u16    u16x16;
typedef __attribute__((ext_vector_type(16))) __bf16 v16bf;
typedef __attribute__((ext_vector_type(8)))  float  v8f;

#define B_  8
#define C_  512
#define H_  8
#define D_  64
#define N_  1024   // 32*32 tokens

// ---------- helpers ----------
__device__ __forceinline__ u16 f2bf(float f) {
    u32 u = __builtin_bit_cast(u32, f);
    if ((u & 0x7F800000u) == 0x7F800000u) return (u16)(u >> 16); // inf/nan: truncate
    u32 r = u + 0x7FFFu + ((u >> 16) & 1u);                      // RNE
    return (u16)(r >> 16);
}
__device__ __forceinline__ u32 fkey(float f) {   // monotonic float->uint
    u32 u = __builtin_bit_cast(u32, f);
    return u ^ ((u & 0x80000000u) ? 0xFFFFFFFFu : 0x80000000u);
}
__device__ __forceinline__ float fkeyinv(u32 k) {
    u32 u = k ^ ((k & 0x80000000u) ? 0x80000000u : 0xFFFFFFFFu);
    return __builtin_bit_cast(float, u);
}
__device__ __forceinline__ v16bf make_frag(u16x8 lo, u16x8 hi) {
    u16x16 t;
#pragma unroll
    for (int i = 0; i < 8; ++i) { t[i] = lo[i]; t[i + 8] = hi[i]; }
    return __builtin_bit_cast(v16bf, t);
}
// A fragment (16x32 bf16, row-major src), k contiguous:
// lane<16 holds K0..15, lane>=16 holds K16..31.
__device__ __forceinline__ v16bf load_a(const u16* base, int k0, int half) {
    return __builtin_bit_cast(v16bf, *(const u16x16*)(base + k0 + (half << 4)));
}
// B fragment (32x16 bf16): lane=col, k contiguous; lanes 0-15 K0..15, 16-31 K16..31.
__device__ __forceinline__ v16bf load_b(const u16* base, int k0, int half) {
    return __builtin_bit_cast(v16bf, *(const u16x16*)(base + k0 + (half << 4)));
}
__device__ __forceinline__ v8f wmma_bf16(v16bf a, v16bf b, v8f c) {
    return __builtin_amdgcn_wmma_f32_16x16x32_bf16(false, a, false, b, (short)0, c,
                                                   false, false);
}
__device__ __forceinline__ float wave_max(float v) {
#pragma unroll
    for (int o = 16; o > 0; o >>= 1) v = fmaxf(v, __shfl_xor(v, o, 32));
    return v;
}
__device__ __forceinline__ float wave_sum(float v) {
#pragma unroll
    for (int o = 16; o > 0; o >>= 1) v += __shfl_xor(v, o, 32);
    return v;
}

// ---------- 1. multi-scale avg+max pooling (stride 1, 'same') ----------
template <int KS>
__device__ __forceinline__ float pool_pair(const float* t, int py, int px) {
    constexpr int P = (KS - 1) / 2;
    float s = 0.f, mx = -3.4e38f;
#pragma unroll
    for (int dy = -P; dy <= P; ++dy)
#pragma unroll
        for (int dx = -P; dx <= P; ++dx) {
            int yy = py + dy, xx = px + dx;
            bool ok = (yy >= 0) & (yy < 32) & (xx >= 0) & (xx < 32);
            float v = ok ? t[yy * 32 + xx] : 0.f;
            s += v;
            mx = ok ? fmaxf(mx, v) : mx;
        }
    return s * (1.f / (KS * KS)) + mx;
}
__global__ void dsca_pool_kernel(const float* __restrict__ y, float* __restrict__ yp) {
    __shared__ float t[1024];
    int bc = blockIdx.x, tid = threadIdx.x;
    const float* src = y + (size_t)bc * 1024;
#pragma unroll
    for (int j = 0; j < 4; ++j) t[tid + 256 * j] = src[tid + 256 * j];
    __syncthreads();
#pragma unroll
    for (int j = 0; j < 4; ++j) {
        int pix = tid + 256 * j, py = pix >> 5, px = pix & 31;
        float acc = pool_pair<3>(t, py, px) + pool_pair<5>(t, py, px) + pool_pair<7>(t, py, px);
        yp[(size_t)bc * 1024 + pix] = acc;
    }
}

// ---------- 2. layernorm stats over channels per (b,n) ----------
__global__ void dsca_ln_stats(const float* __restrict__ yp, float* __restrict__ mean,
                              float* __restrict__ rstd) {
    __shared__ float ps[8][32], ps2[8][32];
    int b = blockIdx.x, n0 = blockIdx.y * 32, tid = threadIdx.x;
    int nl = tid & 31, cg = tid >> 5;
    float s = 0.f, s2 = 0.f;
    for (int c = cg; c < C_; c += 8) {
        float v = yp[((size_t)b * C_ + c) * N_ + n0 + nl];
        s += v; s2 += v * v;
    }
    ps[cg][nl] = s; ps2[cg][nl] = s2;
    __syncthreads();
    if (cg == 0) {
#pragma unroll
        for (int g = 1; g < 8; ++g) { s += ps[g][nl]; s2 += ps2[g][nl]; }
        float mu = s * (1.f / C_);
        float var = s2 * (1.f / C_) - mu * mu;
        mean[b * N_ + n0 + nl] = mu;
        rstd[b * N_ + n0 + nl] = rsqrtf(var + 1e-5f);
    }
}

// ---------- 3. layernorm apply + transpose -> yn bf16 [b][n][c] ----------
__global__ void dsca_ln_apply(const float* __restrict__ yp, const float* __restrict__ mean,
                              const float* __restrict__ rstd, const float* __restrict__ g,
                              const float* __restrict__ beta, u16* __restrict__ yn) {
    __shared__ float t[32][33];
    int n0 = blockIdx.x * 32, c0 = blockIdx.y * 32, b = blockIdx.z;
    int tid = threadIdx.x, jn = tid & 31, ic = tid >> 5;
#pragma unroll
    for (int r = 0; r < 4; ++r) {
        int c = ic + 8 * r;
        t[c][jn] = yp[((size_t)b * C_ + c0 + c) * N_ + n0 + jn];
    }
    __syncthreads();
    int jc = tid & 31, in_ = tid >> 5;
#pragma unroll
    for (int r = 0; r < 4; ++r) {
        int n = in_ + 8 * r;
        float m = mean[b * N_ + n0 + n], rs = rstd[b * N_ + n0 + n];
        float v = (t[jc][n] - m) * rs * g[c0 + jc] + beta[c0 + jc];
        yn[((size_t)b * N_ + n0 + n) * C_ + c0 + jc] = f2bf(v);
    }
}

// ---------- 4. transpose+convert x -> xt bf16 [b][n][c] ----------
__global__ void dsca_xpose_bf16(const float* __restrict__ src, u16* __restrict__ dst) {
    __shared__ float t[32][33];
    int n0 = blockIdx.x * 32, c0 = blockIdx.y * 32, b = blockIdx.z;
    int tid = threadIdx.x, jn = tid & 31, ic = tid >> 5;
#pragma unroll
    for (int r = 0; r < 4; ++r) {
        int c = ic + 8 * r;
        t[c][jn] = src[((size_t)b * C_ + c0 + c) * N_ + n0 + jn];
    }
    __syncthreads();
    int jc = tid & 31, in_ = tid >> 5;
#pragma unroll
    for (int r = 0; r < 4; ++r) {
        int n = in_ + 8 * r;
        dst[((size_t)b * N_ + n0 + n) * C_ + c0 + jc] = f2bf(t[jc][n]);
    }
}

// ---------- 5. weight convert f32 -> bf16 ----------
__global__ void dsca_cvt_bf16(const float* __restrict__ s, u16* __restrict__ d, int n) {
    int i = blockIdx.x * 256 + threadIdx.x;
    if (i < n) d[i] = f2bf(s[i]);
}

// ---------- 6. dynamic-k head ----------
__global__ void dsca_dynk(const float* __restrict__ x, const float* __restrict__ w1,
                          const float* __restrict__ b1, const float* __restrict__ w2,
                          const float* __restrict__ b2, int* __restrict__ kvals) {
    __shared__ float pooled[C_];
    __shared__ float h1[128];
    __shared__ float lg[H_];
    int b = blockIdx.x, tid = threadIdx.x, lane = tid & 31, wave = tid >> 5;
    for (int it = 0; it < 64; ++it) {
        int c = wave + 8 * it;
        const float* xp = x + ((size_t)b * C_ + c) * N_;
        float s = 0.f;
        for (int j = 0; j < 32; ++j) s += xp[lane + 32 * j];
        s = wave_sum(s);
        if (lane == 0) pooled[c] = s * (1.f / N_);
    }
    __syncthreads();
    if (tid < 128) {
        float s = b1[tid];
        const float* wp = w1 + (size_t)tid * C_;
        for (int c = 0; c < C_; ++c) s += pooled[c] * wp[c];
        h1[tid] = fmaxf(s, 0.f);
    }
    __syncthreads();
    if (tid < H_) {
        float s = b2[tid];
        const float* wp = w2 + (size_t)tid * 128;
        for (int j = 0; j < 128; ++j) s += h1[j] * wp[j];
        lg[tid] = s;
    }
    __syncthreads();
    if (tid == 0) {
        float m = lg[0];
#pragma unroll
        for (int i = 1; i < H_; ++i) m = fmaxf(m, lg[i]);
        float e[H_], sum = 0.f;
#pragma unroll
        for (int i = 0; i < H_; ++i) { e[i] = __expf(lg[i] - m); sum += e[i]; }
#pragma unroll
        for (int i = 0; i < H_; ++i) {
            int k = (int)floorf(e[i] / sum * (float)N_);
            k = k < 1 ? 1 : (k > N_ ? N_ : k);
            kvals[b * H_ + i] = k;
        }
    }
}

// ---------- 7. bf16 WMMA GEMM, K=512 fixed: Out[M,ldo] = A[M,512] * W[N,512]^T ----
// Wave tile 16x32 (one A fragment, two accumulators). Fully unrolled k-loop so
// the software pipeline and prefetch guards are branch-free at compile time.
__global__ void dsca_gemm_bf16(const u16* __restrict__ A, const u16* __restrict__ W,
                               u16* __restrict__ Out, int ldo) {
    int tid = threadIdx.x, lane = tid & 31, wave = tid >> 5;
    int m0 = blockIdx.x * 32 + (wave >> 1) * 16;
    int n0 = blockIdx.y * 64 + (wave & 1) * 32;
    int rowl = lane & 15, half = lane >> 4;
    const u16* ap  = A + (size_t)(m0 + rowl) * C_;
    const u16* wp0 = W + (size_t)(n0 + rowl) * C_;
    const u16* wp1 = wp0 + (size_t)16 * C_;
    v8f acc0 = {0.f, 0.f, 0.f, 0.f, 0.f, 0.f, 0.f, 0.f};
    v8f acc1 = acc0;
    v16bf a  = load_a(ap, 0, half);
    v16bf b0 = load_b(wp0, 0, half);
    v16bf b1 = load_b(wp1, 0, half);
#pragma unroll
    for (int k0 = 0; k0 < C_ - 32; k0 += 32) {
        v16bf an  = load_a(ap, k0 + 32, half);
        v16bf b0n = load_b(wp0, k0 + 32, half);
        v16bf b1n = load_b(wp1, k0 + 32, half);
        if (k0 + 128 < C_) {
            __builtin_prefetch(ap + k0 + 128, 0, 0);
            __builtin_prefetch(wp0 + k0 + 128, 0, 0);
            __builtin_prefetch(wp1 + k0 + 128, 0, 0);
        }
        acc0 = wmma_bf16(a, b0, acc0);
        acc1 = wmma_bf16(a, b1, acc1);
        a = an; b0 = b0n; b1 = b1n;
    }
    acc0 = wmma_bf16(a, b0, acc0);
    acc1 = wmma_bf16(a, b1, acc1);
#pragma unroll
    for (int i = 0; i < 8; ++i) {
        int r = m0 + i + (half << 3);
        Out[(size_t)r * ldo + n0 + rowl]      = f2bf(acc0[i]);
        Out[(size_t)r * ldo + n0 + 16 + rowl] = f2bf(acc1[i]);
    }
}

// ---------- 8. transpose v slice of kv -> vT bf16 [b][h*64+dd][m] ----------
__global__ void dsca_vxpose(const u16* __restrict__ kv, u16* __restrict__ vT) {
    __shared__ u16 t[32][68];
    int m0 = blockIdx.x * 32, hd0 = blockIdx.y * 64, b = blockIdx.z;
    int tid = threadIdx.x;
    {
        int dd = tid & 63, im = tid >> 6;
#pragma unroll
        for (int r = 0; r < 8; ++r) {
            int m = im + 4 * r;
            t[m][dd] = kv[((size_t)(b * N_ + m0 + m)) * 1024 + 512 + hd0 + dd];
        }
    }
    __syncthreads();
    {
        int jm = tid & 31, id = tid >> 5;
#pragma unroll
        for (int r = 0; r < 8; ++r) {
            int dd = id + 4 * r;
            vT[((size_t)b * C_ + hd0 + dd) * N_ + m0 + jm] = t[jm][dd];
        }
    }
}

// ---------- 9. flash-style attention w/ exact dynamic top-k threshold ----------
// grid (N/16, H, B), block 128 (4 waves). LDS: scores 16x1024 f32 + radix hist.
__global__ void dsca_attn(const u16* __restrict__ q_lin, const u16* __restrict__ kv_lin,
                          const u16* __restrict__ vT, const int* __restrict__ kvals,
                          const float* __restrict__ temp, u16* __restrict__ attn_out) {
    extern __shared__ char smem[];
    float* sc   = (float*)smem;                      // [16][1024]
    u32*   hist = (u32*)(smem + 65536);              // [4][256]
    float* thr  = (float*)(smem + 65536 + 4096);     // [16]
    float* scl  = thr + 16;                          // [16]
    u32*   wsel = (u32*)(scl + 16);                  // [8]

    int tid = threadIdx.x, lane = tid & 31, wave = tid >> 5;
    int rowl = lane & 15, half = lane >> 4;
    int b = blockIdx.z, h = blockIdx.y, m0q = blockIdx.x * 16;
    float tscale = temp[h];
    int kkeep = kvals[b * H_ + h];

    // ---- phase 1: scores = (q . k^T) * temp, 16 rows x 1024 cols via WMMA ----
    // Each wave covers 16 column tiles; last iteration peeled so the steady
    // state loads the next K-fragments unconditionally (branch-free pipeline).
    const u16* qp = q_lin + ((size_t)(b * N_ + m0q + rowl)) * C_ + h * D_;
    v16bf a0 = load_a(qp, 0, half);
    v16bf a1 = load_a(qp, 32, half);
    {
        const u16* kp0 = kv_lin + ((size_t)(b * N_ + rowl)) * 1024 + h * D_;
        v16bf b0 = load_b(kp0 + (size_t)(wave * 16) * 1024, 0, half);
        v16bf b1 = load_b(kp0 + (size_t)(wave * 16) * 1024, 32, half);
#pragma unroll 5
        for (int it = 0; it < 15; ++it) {
            int ct = wave + it * 4;
            const u16* kpn = kp0 + (size_t)((ct + 4) * 16) * 1024;
            v16bf b0n = load_b(kpn, 0, half);
            v16bf b1n = load_b(kpn, 32, half);
            v8f acc = {0.f, 0.f, 0.f, 0.f, 0.f, 0.f, 0.f, 0.f};
            acc = wmma_bf16(a0, b0, acc);
            acc = wmma_bf16(a1, b1, acc);
            int m0k = ct * 16;
#pragma unroll
            for (int i = 0; i < 8; ++i) {
                int r = i + (half << 3);
                sc[r * 1024 + m0k + rowl] = acc[i] * tscale;
            }
            b0 = b0n; b1 = b1n;
        }
        {   // peeled tail: ct = wave + 60
            v8f acc = {0.f, 0.f, 0.f, 0.f, 0.f, 0.f, 0.f, 0.f};
            acc = wmma_bf16(a0, b0, acc);
            acc = wmma_bf16(a1, b1, acc);
            int m0k = (wave + 60) * 16;
#pragma unroll
            for (int i = 0; i < 8; ++i) {
                int r = i + (half << 3);
                sc[r * 1024 + m0k + rowl] = acc[i] * tscale;
            }
        }
    }
    __syncthreads();

    // ---- phase 2: per-row radix select of k-th largest (exact threshold) ----
    for (int wr = 0; wr < 4; ++wr) {
        int r = wave * 4 + wr;
        u32 prefix = 0;
        int kk = kkeep;
        for (int pass = 3; pass >= 0; --pass) {
#pragma unroll
            for (int z = 0; z < 8; ++z) hist[wave * 256 + lane + 32 * z] = 0u;
            __syncthreads();
            for (int e = 0; e < 32; ++e) {
                u32 u = fkey(sc[r * 1024 + e * 32 + lane]);
                if (((u >> (8 * pass)) >> 8) == ((prefix >> (8 * pass)) >> 8)) {
                    u32 dig = (u >> (8 * pass)) & 255u;
                    atomicAdd(&hist[wave * 256 + dig], 1u);
                }
            }
            __syncthreads();
            if (lane == 0) {
                int rem = kk;
                u32 dig = 0;
                for (int d = 255; d >= 0; --d) {
                    u32 c = hist[wave * 256 + d];
                    if ((u32)rem <= c) { dig = (u32)d; break; }
                    rem -= (int)c;
                }
                wsel[wave * 2] = dig;
                wsel[wave * 2 + 1] = (u32)rem;
            }
            __syncthreads();
            prefix |= wsel[wave * 2] << (8 * pass);
            kk = (int)wsel[wave * 2 + 1];
        }
        if (lane == 0) thr[r] = fkeyinv(prefix);
    }
    __syncthreads();

    // ---- phase 3: masked (zeros, not -inf) softmax; p stored bf16 in place ----
    for (int wr = 0; wr < 4; ++wr) {
        int r = wave * 4 + wr;
        float th = thr[r];
        float mx = -3.4e38f;
        for (int e = 0; e < 32; ++e) {
            float s = sc[r * 1024 + e * 32 + lane];
            float v = (s >= th) ? s : 0.f;
            mx = fmaxf(mx, v);
        }
        mx = wave_max(mx);
        u16* pr = (u16*)(sc + r * 1024);
        float sum = 0.f;
        for (int e = 0; e < 32; ++e) {
            int idx = e * 32 + lane;
            float s = sc[r * 1024 + idx];
            float v = (s >= th) ? s : 0.f;
            float p = __expf(v - mx);
            sum += p;
            pr[idx] = f2bf(p);   // safe: bf16 writes trail the f32 reads in the row
        }
        sum = wave_sum(sum);
        if (lane == 0) scl[r] = 1.f / sum;
    }
    __syncthreads();

    // ---- phase 4: out = softmax(scores) . v via WMMA, 1/sum folded in ----
    {
        int ddt = wave * 16;
        const u16* pbase = (const u16*)sc + rowl * 2048;  // row stride = 2048 bf16
        const u16* vb = vT + ((size_t)b * C_ + h * D_ + ddt + rowl) * N_;
        v8f acc = {0.f, 0.f, 0.f, 0.f, 0.f, 0.f, 0.f, 0.f};
        v16bf bb = load_b(vb, 0, half);
#pragma unroll 8
        for (int k0 = 0; k0 < 992; k0 += 32) {
            v16bf bbn = load_b(vb, k0 + 32, half);       // unconditional next-load
            v16bf a = load_a(pbase, k0, half);           // ds_load_b128 x2
            acc = wmma_bf16(a, bb, acc);
            bb = bbn;
        }
        {   // peeled tail k0 = 992
            v16bf a = load_a(pbase, 992, half);
            acc = wmma_bf16(a, bb, acc);
        }
#pragma unroll
        for (int i = 0; i < 8; ++i) {
            int r = i + (half << 3);
            float o = acc[i] * scl[r];
            attn_out[((size_t)(b * N_ + m0q + r)) * C_ + h * D_ + ddt + rowl] = f2bf(o);
        }
    }
}

// ---------- 10. proj GEMM + bias, write transposed f32 [b][c][n] ----------
__global__ void dsca_proj(const u16* __restrict__ A, const u16* __restrict__ W,
                          const float* __restrict__ bias, float* __restrict__ out) {
    __shared__ float t[32][33];
    int tid = threadIdx.x, lane = tid & 31, wave = tid >> 5;
    int m0 = blockIdx.x * 32, c0 = blockIdx.y * 32;
    int mq = (wave >> 1) * 16, nq = (wave & 1) * 16;
    int rowl = lane & 15, half = lane >> 4;
    const u16* ap = A + (size_t)(m0 + mq + rowl) * C_;
    const u16* wp = W + (size_t)(c0 + nq + rowl) * C_;
    v8f acc = {0.f, 0.f, 0.f, 0.f, 0.f, 0.f, 0.f, 0.f};
    v16bf a = load_a(ap, 0, half);
    v16bf bb = load_b(wp, 0, half);
#pragma unroll
    for (int k0 = 0; k0 < C_ - 32; k0 += 32) {
        v16bf an = load_a(ap, k0 + 32, half);
        v16bf bn = load_b(wp, k0 + 32, half);
        if (k0 + 128 < C_) {
            __builtin_prefetch(ap + k0 + 128, 0, 0);
            __builtin_prefetch(wp + k0 + 128, 0, 0);
        }
        acc = wmma_bf16(a, bb, acc);
        a = an; bb = bn;
    }
    acc = wmma_bf16(a, bb, acc);
#pragma unroll
    for (int i = 0; i < 8; ++i) {
        int nl = mq + i + (half << 3);
        int cl = nq + rowl;
        t[cl][nl] = acc[i] + bias[c0 + cl];
    }
    __syncthreads();
    int b = m0 >> 10, n0 = m0 & 1023;
#pragma unroll
    for (int r = 0; r < 8; ++r) {
        int nl = tid & 31, cl = (tid >> 5) + 4 * r;
        out[((size_t)b * C_ + c0 + cl) * N_ + n0 + nl] = t[cl][nl];
    }
}

// ---------- host-side orchestration ----------
extern "C" void kernel_launch(void* const* d_in, const int* in_sizes, int n_in,
                              void* d_out, int out_size, void* d_ws, size_t ws_size,
                              hipStream_t stream) {
    (void)in_sizes; (void)n_in; (void)out_size; (void)ws_size;
    const float* x      = (const float*)d_in[0];
    const float* y      = (const float*)d_in[1];
    const float* temp   = (const float*)d_in[2];
    const float* q_w    = (const float*)d_in[3];
    const float* kv_w   = (const float*)d_in[4];
    const float* proj_w = (const float*)d_in[5];
    const float* proj_b = (const float*)d_in[6];
    const float* dk_w1  = (const float*)d_in[7];
    const float* dk_b1  = (const float*)d_in[8];
    const float* dk_w2  = (const float*)d_in[9];
    const float* dk_b2  = (const float*)d_in[10];
    const float* ln_g   = (const float*)d_in[11];
    const float* ln_b   = (const float*)d_in[12];
    float* out = (float*)d_out;

    char* ws = (char*)d_ws;
    size_t off = 0;
    auto carve = [&](size_t bytes) { char* p = ws + off; off += (bytes + 255) & ~(size_t)255; return p; };

    float* yp     = (float*)carve((size_t)B_ * C_ * N_ * 4);   // 16 MB
    float* mean   = (float*)carve((size_t)B_ * N_ * 4);
    float* rstd   = (float*)carve((size_t)B_ * N_ * 4);
    u16*   xt     = (u16*)  carve((size_t)B_ * N_ * C_ * 2);   // 8 MB
    u16*   yn     = (u16*)  carve((size_t)B_ * N_ * C_ * 2);   // 8 MB
    u16*   qwb    = (u16*)  carve((size_t)C_ * C_ * 2);
    u16*   kvwb   = (u16*)  carve((size_t)2 * C_ * C_ * 2);
    u16*   pwb    = (u16*)  carve((size_t)C_ * C_ * 2);
    u16*   q_lin  = (u16*)  carve((size_t)B_ * N_ * C_ * 2);   // 8 MB
    u16*   kv_lin = (u16*)  carve((size_t)B_ * N_ * 1024 * 2); // 16 MB
    u16*   vT     = (u16*)  carve((size_t)B_ * C_ * N_ * 2);   // 8 MB
    u16*   aout   = (u16*)  carve((size_t)B_ * N_ * C_ * 2);   // 8 MB
    int*   kvals  = (int*)  carve((size_t)B_ * H_ * 4);

    // 1. pooling
    dsca_pool_kernel<<<B_ * C_, 256, 0, stream>>>(y, yp);
    // 2-3. layernorm (stats + apply/transpose -> bf16)
    dsca_ln_stats<<<dim3(B_, N_ / 32), 256, 0, stream>>>(yp, mean, rstd);
    dsca_ln_apply<<<dim3(N_ / 32, C_ / 32, B_), 256, 0, stream>>>(yp, mean, rstd, ln_g, ln_b, yn);
    // 4. x -> [b,n,c] bf16
    dsca_xpose_bf16<<<dim3(N_ / 32, C_ / 32, B_), 256, 0, stream>>>(x, xt);
    // 5. weight conversions
    dsca_cvt_bf16<<<(C_ * C_ + 255) / 256, 256, 0, stream>>>(q_w, qwb, C_ * C_);
    dsca_cvt_bf16<<<(2 * C_ * C_ + 255) / 256, 256, 0, stream>>>(kv_w, kvwb, 2 * C_ * C_);
    dsca_cvt_bf16<<<(C_ * C_ + 255) / 256, 256, 0, stream>>>(proj_w, pwb, C_ * C_);
    // 6. dynamic-k head
    dsca_dynk<<<B_, 256, 0, stream>>>(x, dk_w1, dk_b1, dk_w2, dk_b2, kvals);
    // 7. projections (WMMA, wave tile 16x32, fully unrolled pipeline)
    dsca_gemm_bf16<<<dim3(B_ * N_ / 32, C_ / 64), 128, 0, stream>>>(xt, qwb, q_lin, C_);
    dsca_gemm_bf16<<<dim3(B_ * N_ / 32, 1024 / 64), 128, 0, stream>>>(yn, kvwb, kv_lin, 1024);
    // 8. v transpose for contiguous B-fragments
    dsca_vxpose<<<dim3(N_ / 32, C_ / 64, B_), 256, 0, stream>>>(kv_lin, vT);
    // 9. attention (scores WMMA -> radix top-k -> masked softmax -> out WMMA)
    size_t attn_smem = 65536 + 4096 + 64 + 64 + 64;
    dsca_attn<<<dim3(N_ / 16, H_, B_), 128, attn_smem, stream>>>(q_lin, kv_lin, vT, kvals, temp, aout);
    // 10. output projection + bias, transposed store
    dsca_proj<<<dim3(B_ * N_ / 32, C_ / 32), 128, 0, stream>>>(aout, pwb, proj_b, out);
}
